// HierarchicalPointerHeadV2_55980603736508
// MI455X (gfx1250) — matrix operand order repaired and checked by reading, not compiled, not run on hardware
//
#include <hip/hip_runtime.h>
#include <hip/hip_bf16.h>
#include <math.h>

// ---------- shapes ----------
#define Bq 8
#define Tq 64
#define Dq 1024
#define Sq 32
#define Eq 16
#define Vq 50257

typedef __attribute__((ext_vector_type(16))) __bf16 v16bf;
typedef __attribute__((ext_vector_type(8)))  __bf16 v8bf;
typedef __attribute__((ext_vector_type(4)))  __bf16 v4bf;
typedef __attribute__((ext_vector_type(8)))  float  v8f;

// ---------------------------------------------------------------------------
// Zero-fill of pointer_probs (103MB) — pure streaming stores, float4 wide.
// ---------------------------------------------------------------------------
__global__ __launch_bounds__(256) void zero_kernel(float4* __restrict__ p, long n4) {
    long i = (long)blockIdx.x * blockDim.x + threadIdx.x;
    long stride = (long)gridDim.x * blockDim.x;
    float4 z; z.x = 0.f; z.y = 0.f; z.z = 0.f; z.w = 0.f;
    for (; i < n4; i += stride) p[i] = z;
}

// ---------------------------------------------------------------------------
// C[M,N] = A[M,K] @ W[K,N] + bias   via v_wmma_f32_16x16x32_bf16.
// Block = 128 threads (4 waves), tile BM=64 BN=64 BK=32, double-buffered LDS.
// Wave w owns rows [64*by + 16w, +16) across all 64 cols (4 accumulators).
// LDS: As row-major [64][32] bf16; Bs transposed [n=64][k=32] bf16 so the
// B-fragment (two runs of 8 K per lane per ISA 7.12.2) is two 16B LDS loads.
// Staging loops are fully unrolled constant-stride loads (trip count 4).
// ---------------------------------------------------------------------------
__global__ __launch_bounds__(128) void gemm_bias_wmma(
    const float* __restrict__ A, const float* __restrict__ W,
    const float* __restrict__ bias, float* __restrict__ C,
    int K, int N)
{
    __shared__ __bf16 As[2][64 * 32];
    __shared__ __bf16 Bs[2][64 * 32];

    const int tid  = threadIdx.x;
    const int wave = tid >> 5;
    const int lane = tid & 31;
    const int m0   = blockIdx.y * 64;
    const int n0   = blockIdx.x * 64;

    v8f acc[4] = {{}, {}, {}, {}};

    // ---- per-thread staging geometry (constant across the k loop) ----
    // A tile: thread owns rows (tid>>3)+16*it, 4 consecutive k at (tid&7)*4
    const int arow = tid >> 3;          // 0..15
    const int acol = (tid & 7) * 4;     // 0,4,..,28
    // W tile: thread owns k rows (tid>>4)+8*it, 4 consecutive n at (tid&15)*4
    const int brow = tid >> 4;          // 0..7
    const int bcol = (tid & 15) * 4;    // 0,4,..,60
    const float* pA = A + (size_t)(m0 + arow) * K + acol;
    const float* pW = W + (size_t)brow * N + n0 + bcol;

    // ---- WMMA fragment addressing (ISA 7.12.2 16-bit layouts) ----
    // A frag: lane L -> row L&15; lanes<16 take K {0..7,16..23}, >=16 {8..15,24..31}
    const int mloc  = wave * 16 + (lane & 15);
    const int kA    = (lane >> 4) * 8;     // 0 or 8
    const int koffB = (lane >> 4) * 8;     // 0 or 8 (B frag K {0..7,16..23} / {8..15,24..31})

    const int nK = K / 32;
    float4 ra[4], rb[4];

    // ---- preload tile 0 ----
#pragma unroll
    for (int it = 0; it < 4; ++it) {
        ra[it] = *(const float4*)(pA + (size_t)(16 * it) * K);
        rb[it] = *(const float4*)(pW + (size_t)(8 * it) * N);
    }
#pragma unroll
    for (int it = 0; it < 4; ++it) {
        v4bf a4;
        a4[0] = (__bf16)ra[it].x; a4[1] = (__bf16)ra[it].y;
        a4[2] = (__bf16)ra[it].z; a4[3] = (__bf16)ra[it].w;
        *(v4bf*)(&As[0][(arow + 16 * it) * 32 + acol]) = a4;
        const int kidx = brow + 8 * it;
        Bs[0][(bcol + 0) * 32 + kidx] = (__bf16)rb[it].x;
        Bs[0][(bcol + 1) * 32 + kidx] = (__bf16)rb[it].y;
        Bs[0][(bcol + 2) * 32 + kidx] = (__bf16)rb[it].z;
        Bs[0][(bcol + 3) * 32 + kidx] = (__bf16)rb[it].w;
    }
    __syncthreads();

    for (int kt = 0; kt < nK; ++kt) {
        const int cur = kt & 1;
        const int nxt = cur ^ 1;
        const bool more = (kt + 1) < nK;

        // 1) issue global loads for tile kt+1 (in flight during WMMA below)
        if (more) {
            const float* pA2 = pA + (kt + 1) * 32;
            const float* pW2 = pW + (size_t)((kt + 1) * 32) * N;
#pragma unroll
            for (int it = 0; it < 4; ++it) {
                ra[it] = *(const float4*)(pA2 + (size_t)(16 * it) * K);
                rb[it] = *(const float4*)(pW2 + (size_t)(8 * it) * N);
            }
            if (kt + 2 < nK) {
                __builtin_prefetch(pA + (kt + 2) * 32, 0, 0);
                __builtin_prefetch(pW + (size_t)((kt + 2) * 32) * N, 0, 0);
            }
        }

        // 2) fragments + WMMA from current buffer
        v8bf alo = *(const v8bf*)(&As[cur][mloc * 32 + kA]);
        v8bf ahi = *(const v8bf*)(&As[cur][mloc * 32 + 16 + kA]);
        v16bf a = __builtin_shufflevector(alo, ahi,
                    0, 1, 2, 3, 4, 5, 6, 7, 8, 9, 10, 11, 12, 13, 14, 15);
#pragma unroll
        for (int nt = 0; nt < 4; ++nt) {
            const int n = nt * 16 + (lane & 15);
            v8bf blo = *(const v8bf*)(&Bs[cur][n * 32 + koffB]);
            v8bf bhi = *(const v8bf*)(&Bs[cur][n * 32 + 16 + koffB]);
            v16bf b = __builtin_shufflevector(blo, bhi,
                    0, 1, 2, 3, 4, 5, 6, 7, 8, 9, 10, 11, 12, 13, 14, 15);
            acc[nt] = __builtin_amdgcn_wmma_f32_16x16x32_bf16(
                false, a, false, b, (short)0, acc[nt], false, false);
        }

        // 3) stage tile kt+1 into the other buffer
        if (more) {
#pragma unroll
            for (int it = 0; it < 4; ++it) {
                v4bf a4;
                a4[0] = (__bf16)ra[it].x; a4[1] = (__bf16)ra[it].y;
                a4[2] = (__bf16)ra[it].z; a4[3] = (__bf16)ra[it].w;
                *(v4bf*)(&As[nxt][(arow + 16 * it) * 32 + acol]) = a4;
                const int kidx = brow + 8 * it;
                Bs[nxt][(bcol + 0) * 32 + kidx] = (__bf16)rb[it].x;
                Bs[nxt][(bcol + 1) * 32 + kidx] = (__bf16)rb[it].y;
                Bs[nxt][(bcol + 2) * 32 + kidx] = (__bf16)rb[it].z;
                Bs[nxt][(bcol + 3) * 32 + kidx] = (__bf16)rb[it].w;
            }
        }
        __syncthreads();
    }

    // writeback (ISA 32-bit C/D layout): VGPR i -> row (L>=16 ? 8+i : i)
    const int rowbase = m0 + wave * 16 + (lane >> 4) * 8;
#pragma unroll
    for (int nt = 0; nt < 4; ++nt) {
        const int col  = n0 + nt * 16 + (lane & 15);
        const float bv = bias[col];
#pragma unroll
        for (int i = 0; i < 8; ++i)
            C[(size_t)(rowbase + i) * N + col] = acc[nt][i] + bv;
    }
}

// ---------------------------------------------------------------------------
// scene_dot[b,s] = scene[b,s,:] . Wp[D:2D]   (one wave per (b,s))
// ---------------------------------------------------------------------------
__global__ __launch_bounds__(32) void scene_dot_kernel(
    const float* __restrict__ scene, const float* __restrict__ Wp,
    float* __restrict__ sdot)
{
    const int bs = blockIdx.x, lane = threadIdx.x;
    const float4* srow = (const float4*)(scene + (size_t)bs * Dq);
    const float4* wph  = (const float4*)(Wp + Dq);
    float acc = 0.f;
#pragma unroll
    for (int j = 0; j < 8; ++j) {
        float4 s = srow[lane + 32 * j];
        float4 w = wph[lane + 32 * j];
        acc += s.x * w.x + s.y * w.y + s.z * w.z + s.w * w.w;
    }
    for (int o = 16; o; o >>= 1) acc += __shfl_xor(acc, o, 32);
    if (lane == 0) sdot[bs] = acc;
}

// ---------------------------------------------------------------------------
// One wave per (b,t): scores over S=32 (lane == s), softmax via shuffles,
// fused p_gen = sigmoid((dec.Wp_lo + sum_s attn_s*scene_dot[b,s] + bp-0.5)*10)
// ---------------------------------------------------------------------------
__global__ __launch_bounds__(32) void attn_pgen_kernel(
    const float* __restrict__ q, const float* __restrict__ k,
    const float* __restrict__ sdot, const float* __restrict__ dec,
    const float* __restrict__ Wp, const float* __restrict__ bp,
    float* __restrict__ attn, float* __restrict__ p_gen)
{
    const int bt = blockIdx.x;          // b*T + t
    const int b  = bt >> 6;
    const int lane = threadIdx.x;       // == s
    const float4* qrow = (const float4*)(q + (size_t)bt * Dq);
    const float4* krow = (const float4*)(k + (size_t)(b * Sq + lane) * Dq);
    const float4* drow = (const float4*)(dec + (size_t)bt * Dq);
    const float4* wplo = (const float4*)Wp;

    float dot = 0.f, dq = 0.f;
    for (int i = 0; i < Dq / 4; ++i) {
        float4 qa = qrow[i], kb = krow[i];
        dot += qa.x * kb.x + qa.y * kb.y + qa.z * kb.z + qa.w * kb.w;
    }
#pragma unroll
    for (int j = 0; j < 8; ++j) {
        float4 d = drow[lane + 32 * j];
        float4 w = wplo[lane + 32 * j];
        dq += d.x * w.x + d.y * w.y + d.z * w.z + d.w * w.w;
    }

    float score = dot * 0.03125f;       // 1/sqrt(1024)
    float mx = score;
    for (int o = 16; o; o >>= 1) mx = fmaxf(mx, __shfl_xor(mx, o, 32));
    float e = __expf(score - mx);
    float sum = e;
    for (int o = 16; o; o >>= 1) sum += __shfl_xor(sum, o, 32);
    float a = e / sum;
    attn[bt * Sq + lane] = a;

    float g = a * sdot[b * Sq + lane];
    for (int o = 16; o; o >>= 1) g  += __shfl_xor(g, o, 32);
    for (int o = 16; o; o >>= 1) dq += __shfl_xor(dq, o, 32);
    if (lane == 0) {
        float x = (dq + g + bp[0] - 0.5f) * 10.0f;
        p_gen[bt] = 1.0f / (1.0f + __expf(-x));
    }
}

// ---------------------------------------------------------------------------
// logits[b,s,e] = dec_last[b] . emb[ent_ids[b,s,e]]   (one wave per (b,s,e))
// ---------------------------------------------------------------------------
__global__ __launch_bounds__(32) void ent_logit_kernel(
    const float* __restrict__ dec, const float* __restrict__ emb,
    const int* __restrict__ ids, float* __restrict__ logits)
{
    const int g = blockIdx.x;           // b*S*E + s*E + e
    const int b = g >> 9;
    const int lane = threadIdx.x;
    const int id = ids[g];
    const float4* dl = (const float4*)(dec + ((size_t)b * Tq + (Tq - 1)) * Dq);
    const float4* em = (const float4*)(emb + (size_t)id * Dq);
    float acc = 0.f;
#pragma unroll
    for (int j = 0; j < 8; ++j) {
        float4 d = dl[lane + 32 * j];
        float4 e = em[lane + 32 * j];
        acc += d.x * e.x + d.y * e.y + d.z * e.z + d.w * e.w;
    }
    for (int o = 16; o; o >>= 1) acc += __shfl_xor(acc, o, 32);
    if (lane == 0) logits[g] = acc;
}

// ---------------------------------------------------------------------------
// salience[b,s,:] = softmax over E=16 (one wave per (b,s); lanes>=16 idle)
// ---------------------------------------------------------------------------
__global__ __launch_bounds__(32) void salience_kernel(
    const float* __restrict__ logits, float* __restrict__ sal)
{
    const int bs = blockIdx.x, lane = threadIdx.x;
    float x = (lane < Eq) ? logits[bs * Eq + lane] : -3.4e38f;
    float mx = x;
    for (int o = 16; o; o >>= 1) mx = fmaxf(mx, __shfl_xor(mx, o, 32));
    float e = (lane < Eq) ? __expf(x - mx) : 0.f;
    float sum = e;
    for (int o = 16; o; o >>= 1) sum += __shfl_xor(sum, o, 32);
    if (lane < Eq) sal[bs * Eq + lane] = e / sum;
}

// ---------------------------------------------------------------------------
// Sparse scatter: out[b,t, ids[b,s,e]] += attn[b,t,s] * sal[b,s,e]
// one thread per (b,t,s,e); atomic f32 add (no-return, L2-resident)
// ---------------------------------------------------------------------------
__global__ __launch_bounds__(256) void scatter_kernel(
    const float* __restrict__ attn, const float* __restrict__ sal,
    const int* __restrict__ ids, float* __restrict__ pp)
{
    const int gid = blockIdx.x * 256 + threadIdx.x;   // < B*T*S*E = 262144
    const int bt = gid >> 9;                          // b*T + t
    const int se = gid & 511;                         // s*E + e
    const int b  = bt >> 6;
    const float val = attn[bt * Sq + (se >> 4)] * sal[b * 512 + se];
    const int v = ids[b * 512 + se];
    atomicAdd(pp + (size_t)bt * Vq + v, val);
}

// ---------------------------------------------------------------------------
extern "C" void kernel_launch(void* const* d_in, const int* in_sizes, int n_in,
                              void* d_out, int out_size, void* d_ws, size_t ws_size,
                              hipStream_t stream) {
    const float* dec   = (const float*)d_in[0];
    const float* scene = (const float*)d_in[1];
    const float* emb   = (const float*)d_in[2];
    const int*   ids   = (const int*)  d_in[3];
    const float* Wq    = (const float*)d_in[4];
    const float* bq    = (const float*)d_in[5];
    const float* Wk    = (const float*)d_in[6];
    const float* bk    = (const float*)d_in[7];
    const float* Wp    = (const float*)d_in[8];
    const float* bp    = (const float*)d_in[9];

    float* p_gen = (float*)d_out;               // (B,T,1) = 512
    float* pp    = (float*)d_out + Bq * Tq;     // (B,T,V)

    float* ws     = (float*)d_ws;
    float* q      = ws;                          // 512*1024
    float* kproj  = ws + 524288;                 // 256*1024
    float* attn   = ws + 786432;                 // 8*64*32
    float* sdot   = ws + 802816;                 // 8*32
    float* logits = ws + 803072;                 // 8*32*16
    float* sal    = ws + 807168;                 // 8*32*16

    const long n4 = (long)Bq * Tq * Vq / 4;      // pointer_probs float4 count
    zero_kernel<<<2048, 256, 0, stream>>>((float4*)pp, n4);

    // q = dec @ Wq + bq  (M=512)   k = scene @ Wk + bk  (M=256)
    gemm_bias_wmma<<<dim3(Dq / 64, (Bq * Tq) / 64), 128, 0, stream>>>(dec, Wq, bq, q, Dq, Dq);
    gemm_bias_wmma<<<dim3(Dq / 64, (Bq * Sq) / 64), 128, 0, stream>>>(scene, Wk, bk, kproj, Dq, Dq);

    scene_dot_kernel<<<Bq * Sq, 32, 0, stream>>>(scene, Wp, sdot);
    attn_pgen_kernel<<<Bq * Tq, 32, 0, stream>>>(q, kproj, sdot, dec, Wp, bp, attn, p_gen);

    ent_logit_kernel<<<Bq * Sq * Eq, 32, 0, stream>>>(dec, emb, ids, logits);
    salience_kernel<<<Bq * Sq, 32, 0, stream>>>(logits, sal);

    scatter_kernel<<<(Bq * Tq * Sq * Eq) / 256, 256, 0, stream>>>(attn, sal, ids, pp);
}
